// PreNormedHGTCavAttention_89489938580206
// MI455X (gfx1250) — matrix operand order repaired and checked by reading, not compiled, not run on hardware
//
#include <hip/hip_runtime.h>

// ---------------------------------------------------------------------------
// PreNormedHGTCavAttention fused kernel for gfx1250 (MI455X, wave32, WMMA)
// L=5 tokens, H=32, W=128, C=256, HEADS=8, DH=32, NUM_TYPES=2, NUM_REL=4
// One workgroup = 16 pixels, 256 threads (8 waves). All intermediates in LDS.
// GEMMs use v_wmma_f32_16x16x32_f16 (f16 in, f32 accumulate).
// ---------------------------------------------------------------------------

typedef __attribute__((ext_vector_type(16))) _Float16 v16h;
typedef __attribute__((ext_vector_type(8)))  float    v8f;

#define LTOK   5
#define HEADS  8
#define DH     32
#define CCH    256
#define NPIX   4096          // 32*128
#define PTILE  16            // pixels per workgroup
#define RS     264           // padded LDS row stride in halves (256 + 8 pad)

// LDS region offsets (in halves). Regions: Q, K, V, and XN (reused as OUT_HEAD).
#define OQ   0
#define OK_  (80 * RS)       // 80 rows = 5 tokens * 16 pixels
#define OV   (160 * RS)
#define OXN  (240 * RS)      // xn in phases 0-1, out_head in phases 2-3
#define SMEM_HALVES (320 * RS)
#define SMEM_BYTES  (SMEM_HALVES * 2)   // 168,960 bytes

__device__ __forceinline__ v8f wmma16(v16h a, v16h b, v8f c) {
  return __builtin_amdgcn_wmma_f32_16x16x32_f16(
      /*neg_a=*/false, a, /*neg_b=*/false, b,
      /*c_mod=*/(short)0, c, /*reuse_a=*/false, /*reuse_b=*/false);
}

// A fragment (16xK tile, K-step of 32) from padded LDS f16 buffer.
// Row (M) = l*16 + lane%16 within the region; element e -> K = e + 16*(lane/16).
__device__ __forceinline__ v16h load_a_lds(const _Float16* region, int l,
                                           int colBase, int lane) {
  v16h a;
  const int rowBase = (l * PTILE + (lane & 15)) * RS;
  const int hl = (lane >> 4) << 4;   // 0 or 16
#pragma unroll
  for (int e = 0; e < 16; e += 2) {
    int k = hl + e;
    a[e]     = region[rowBase + colBase + k];
    a[e + 1] = region[rowBase + colBase + k + 1];
  }
  return a;
}

// B fragment (32x16 tile) from a row-major f32 weight matrix w[N][rowStrideF],
// with B(K,N) = w[n][kBase + K]. Element e -> K = e + 16*(lane/16).
__device__ __forceinline__ v16h load_b_gw(const float* w, int nBase,
                                          int kBase, int lane) {
  v16h b;
  const int n  = nBase + (lane & 15);
  const int kb = kBase + ((lane >> 4) << 4);
#pragma unroll
  for (int e = 0; e < 16; e += 2) {
    const float2 p = *reinterpret_cast<const float2*>(w + (size_t)n * CCH + kb + e);
    b[e]     = (_Float16)p.x;
    b[e + 1] = (_Float16)p.y;
  }
  return b;
}

// B fragment from a 32x32 relation matrix Wr[p][q] with B(K=p, N=q).
__device__ __forceinline__ v16h load_b_rel(const float* wr, int nBase, int lane) {
  v16h b;
  const int n  = nBase + (lane & 15);
  const int kb = (lane >> 4) << 4;
#pragma unroll
  for (int e = 0; e < 16; ++e) {
    b[e] = (_Float16)wr[(kb + e) * 32 + n];
  }
  return b;
}

__global__ __launch_bounds__(256) void hgt_cav_attention_kernel(
    const float* __restrict__ x,      const int* __restrict__ mask,
    const float* __restrict__ prior,  const float* __restrict__ ln_g,
    const float* __restrict__ ln_b,
    const float* __restrict__ qw, const float* __restrict__ qb,
    const float* __restrict__ kw, const float* __restrict__ kb,
    const float* __restrict__ vw, const float* __restrict__ vb,
    const float* __restrict__ aw, const float* __restrict__ ab,
    const float* __restrict__ ratt, const float* __restrict__ rmsg,
    float* __restrict__ out) {
  extern __shared__ _Float16 smem[];

  const int tid  = threadIdx.x;
  const int lane = tid & 31;
  const int wave = tid >> 5;
  const int pix0 = blockIdx.x * PTILE;

  // node types from prior_encoding[l,0,0,2]; clip at 0 (values are 0/1)
  int ty[LTOK];
#pragma unroll
  for (int l = 0; l < LTOK; ++l) {
    int t = (int)prior[(size_t)l * (32 * 128 * 3) + 2];
    ty[l] = t < 0 ? 0 : t;
  }

  // -------------------------------------------------------------------------
  // Phase 0: load x tile + LayerNorm -> XN (f16, padded rows). One wave / row.
  // -------------------------------------------------------------------------
  for (int r = wave; r < LTOK * PTILE; r += 8) {
    const int l  = r >> 4;
    const int px = r & 15;
    const float* xr = x + ((size_t)l * NPIX + pix0 + px) * CCH + lane * 8;
    const float4 p0 = reinterpret_cast<const float4*>(xr)[0];
    const float4 p1 = reinterpret_cast<const float4*>(xr)[1];
    float vals[8] = {p0.x, p0.y, p0.z, p0.w, p1.x, p1.y, p1.z, p1.w};
    float s = 0.f, ss = 0.f;
#pragma unroll
    for (int u = 0; u < 8; ++u) { s += vals[u]; ss += vals[u] * vals[u]; }
#pragma unroll
    for (int off = 1; off < 32; off <<= 1) {
      s  += __shfl_xor(s, off, 32);
      ss += __shfl_xor(ss, off, 32);
    }
    const float mu   = s * (1.f / 256.f);
    const float var  = ss * (1.f / 256.f) - mu * mu;
    const float rstd = rsqrtf(var + 1e-5f);
    const float4 g0 = reinterpret_cast<const float4*>(ln_g + lane * 8)[0];
    const float4 g1 = reinterpret_cast<const float4*>(ln_g + lane * 8)[1];
    const float4 b0 = reinterpret_cast<const float4*>(ln_b + lane * 8)[0];
    const float4 b1 = reinterpret_cast<const float4*>(ln_b + lane * 8)[1];
    const float gg[8] = {g0.x, g0.y, g0.z, g0.w, g1.x, g1.y, g1.z, g1.w};
    const float bb[8] = {b0.x, b0.y, b0.z, b0.w, b1.x, b1.y, b1.z, b1.w};
    _Float16* xnrow = smem + OXN + r * RS + lane * 8;
#pragma unroll
    for (int u = 0; u < 8; ++u) {
      xnrow[u] = (_Float16)((vals[u] - mu) * rstd * gg[u] + bb[u]);
    }
  }
  __syncthreads();

  // -------------------------------------------------------------------------
  // Phase 1: typed Q/K/V projections. 240 tile-columns over 8 waves.
  // out[px, n] = sum_c xn[px,c] * w[type][n][c] + bias[type][n]
  // -------------------------------------------------------------------------
  for (int tc = wave; tc < 240; tc += 8) {
    const int proj = tc / 80;
    const int rem  = tc % 80;
    const int l    = rem >> 4;
    const int nt   = rem & 15;
    const int t    = ty[l];
    const float* W  = (proj == 0) ? qw : ((proj == 1) ? kw : vw);
    const float* Bv = (proj == 0) ? qb : ((proj == 1) ? kb : vb);
    _Float16* dst = smem + ((proj == 0) ? OQ : ((proj == 1) ? OK_ : OV));
    const float* Wt = W + (size_t)t * (CCH * CCH);
    v8f acc = {};
#pragma unroll
    for (int ks = 0; ks < 8; ++ks) {
      v16h a = load_a_lds(smem + OXN, l, ks * 32, lane);
      v16h b = load_b_gw(Wt, nt * 16, ks * 32, lane);
      acc = wmma16(a, b, acc);
    }
    const float bias = Bv[t * CCH + nt * 16 + (lane & 15)];
#pragma unroll
    for (int v = 0; v < 8; ++v) {
      const int row = l * PTILE + v + ((lane >> 4) << 3);
      dst[row * RS + nt * 16 + (lane & 15)] = (_Float16)(acc[v] + bias);
    }
  }
  __syncthreads();

  // -------------------------------------------------------------------------
  // Phase 2: attention. Each wave owns a (head, i) pair: scores for all j,
  // masked softmax in registers, then message accumulation -> OUT_HEAD (=OXN).
  // -------------------------------------------------------------------------
  for (int task = wave; task < HEADS * LTOK; task += 8) {
    const int head = task / LTOK;
    const int i    = task % LTOK;
    const v8f zero = {};
    v16h aQ = load_a_lds(smem + OQ, i, head * DH, lane);

    float att[LTOK][8];
#pragma unroll
    for (int j = 0; j < LTOK; ++j) {
      const int rel = ty[i] * 2 + ty[j];
      const float* wb = ratt + (size_t)(rel * HEADS + head) * (DH * DH);
      v16h b0 = load_b_rel(wb, 0, lane);
      v16h b1 = load_b_rel(wb, 16, lane);
      v8f d0 = wmma16(aQ, b0, zero);   // qw columns 0..15
      v8f d1 = wmma16(aQ, b1, zero);   // qw columns 16..31
#pragma unroll
      for (int v = 0; v < 8; ++v) {
        const int px = v + ((lane >> 4) << 3);
        const _Float16* krow = smem + OK_ + (j * PTILE + px) * RS + head * DH;
        float p = d0[v] * (float)krow[lane & 15] +
                  d1[v] * (float)krow[16 + (lane & 15)];
        // reduce qw.k over the 16 lanes of this half-wave
        p += __shfl_xor(p, 1, 32);
        p += __shfl_xor(p, 2, 32);
        p += __shfl_xor(p, 4, 32);
        p += __shfl_xor(p, 8, 32);
        att[j][v] = p * 0.17677669529663687f;  // * DH^-0.5
      }
    }
    // masked softmax over j (per pixel, in registers)
#pragma unroll
    for (int v = 0; v < 8; ++v) {
      const int pix = pix0 + v + ((lane >> 4) << 3);
      float sc[LTOK];
      float mx = -3.4e38f;
#pragma unroll
      for (int j = 0; j < LTOK; ++j) {
        const int mv = mask[(size_t)pix * LTOK + j];
        sc[j] = (mv == 0) ? -1e9f : att[j][v];
        mx = fmaxf(mx, sc[j]);
      }
      float sum = 0.f;
#pragma unroll
      for (int j = 0; j < LTOK; ++j) { sc[j] = __expf(sc[j] - mx); sum += sc[j]; }
      const float inv = 1.f / sum;
#pragma unroll
      for (int j = 0; j < LTOK; ++j) att[j][v] = sc[j] * inv;
    }
    // message: out_i[px,c] = sum_j att * (v_j @ w_msg[i,j])[px,c]
    v8f acc0 = {}, acc1 = {};
#pragma unroll
    for (int j = 0; j < LTOK; ++j) {
      const int rel = ty[i] * 2 + ty[j];
      const float* wb = rmsg + (size_t)(rel * HEADS + head) * (DH * DH);
      v16h aV = load_a_lds(smem + OV, j, head * DH, lane);
      v16h b0 = load_b_rel(wb, 0, lane);
      v16h b1 = load_b_rel(wb, 16, lane);
      v8f d0 = wmma16(aV, b0, zero);
      v8f d1 = wmma16(aV, b1, zero);
#pragma unroll
      for (int v = 0; v < 8; ++v) {
        acc0[v] += att[j][v] * d0[v];
        acc1[v] += att[j][v] * d1[v];
      }
    }
#pragma unroll
    for (int v = 0; v < 8; ++v) {
      const int row = i * PTILE + v + ((lane >> 4) << 3);
      _Float16* orow = smem + OXN + row * RS + head * DH + (lane & 15);
      orow[0]  = (_Float16)acc0[v];
      orow[16] = (_Float16)acc1[v];
    }
  }
  __syncthreads();

  // -------------------------------------------------------------------------
  // Phase 3: typed output projection. out[px,c] = sum_i oh[px,i]*aw[t][c][i]+ab
  // -------------------------------------------------------------------------
  for (int tc = wave; tc < 80; tc += 8) {
    const int l  = tc >> 4;
    const int nt = tc & 15;
    const int t  = ty[l];
    const float* Wt = aw + (size_t)t * (CCH * CCH);
    v8f acc = {};
#pragma unroll
    for (int ks = 0; ks < 8; ++ks) {
      v16h a = load_a_lds(smem + OXN, l, ks * 32, lane);
      v16h b = load_b_gw(Wt, nt * 16, ks * 32, lane);
      acc = wmma16(a, b, acc);
    }
    const float bias = ab[t * CCH + nt * 16 + (lane & 15)];
#pragma unroll
    for (int v = 0; v < 8; ++v) {
      const int px = v + ((lane >> 4) << 3);
      out[((size_t)l * NPIX + pix0 + px) * CCH + nt * 16 + (lane & 15)] =
          acc[v] + bias;
    }
  }
}

extern "C" void kernel_launch(void* const* d_in, const int* in_sizes, int n_in,
                              void* d_out, int out_size, void* d_ws, size_t ws_size,
                              hipStream_t stream) {
  (void)in_sizes; (void)n_in; (void)out_size; (void)d_ws; (void)ws_size;
  const float* x     = (const float*)d_in[0];
  const int*   mask  = (const int*)d_in[1];
  const float* prior = (const float*)d_in[2];
  const float* ln_g  = (const float*)d_in[3];
  const float* ln_b  = (const float*)d_in[4];
  const float* qw    = (const float*)d_in[5];
  const float* qb    = (const float*)d_in[6];
  const float* kw    = (const float*)d_in[7];
  const float* kb    = (const float*)d_in[8];
  const float* vw    = (const float*)d_in[9];
  const float* vb    = (const float*)d_in[10];
  const float* aw    = (const float*)d_in[11];
  const float* ab    = (const float*)d_in[12];
  const float* ratt  = (const float*)d_in[13];
  const float* rmsg  = (const float*)d_in[14];
  float* out = (float*)d_out;

  dim3 grid(NPIX / PTILE);   // 256 workgroups
  dim3 block(256);           // 8 waves of 32
  hgt_cav_attention_kernel<<<grid, block, SMEM_BYTES, stream>>>(
      x, mask, prior, ln_g, ln_b, qw, qb, kw, kb, vw, vb, aw, ab, ratt, rmsg,
      out);
}
